// Evolution_with_edge_att_no_share_67860483276945
// MI455X (gfx1250) — matrix operand order repaired and checked by reading, not compiled, not run on hardware
//
#include <hip/hip_runtime.h>
#include <hip/hip_bf16.h>

// CDNA5 / gfx1250: wave32, WMMA bf16 16x16x32 with fp32 accumulation.
typedef __attribute__((ext_vector_type(16))) __bf16 v16bf;
typedef __attribute__((ext_vector_type(8)))  float  v8f;

#define EPSN 1e-5f

__device__ __forceinline__ unsigned short f2bf(float f) {
  unsigned int u = __float_as_uint(f);
  unsigned int r = (u + 0x7FFFu + ((u >> 16) & 1u)) >> 16;  // RNE
  return (unsigned short)r;
}
__device__ __forceinline__ float bf2f(unsigned short h) {
  return __uint_as_float(((unsigned int)h) << 16);
}

union FragU { int4 q[2]; v16bf v; };

// ---------------------------------------------------------------------------
// Generic per-polygon GEMM:  Out[m,n] = sum_k W[m,k] * im2col(X)[k,n]
//   TAPS==9 : circular dilated conv1d (im2col built from LDS-staged X)
//   TAPS==1 : plain 1x1 GEMM over channels (panel built straight from global)
// One block = one polygon (n in 0..127 points), 8 wave32 per block.
// Double-buffered LDS im2col panel (1 barrier/chunk) + register-pipelined A.
// flags: 1=relu, 2=bn(after relu), 4=residual(+X), 8=extra per-poly row vec
// ---------------------------------------------------------------------------
template<int COUTB, int TAPS>
__global__ __launch_bounds__(256)
void snake_gemm(const unsigned short* __restrict__ Xbase, int xPolyStride,
                int Cin, int dil, int Kpad,
                const unsigned short* __restrict__ Wb,
                const float* __restrict__ bias,
                const float* __restrict__ bng, const float* __restrict__ bnb,
                const float* __restrict__ bnm, const float* __restrict__ bnv,
                const float* __restrict__ extra, int extraStride,
                int flags,
                unsigned short* __restrict__ Out, int outPolyStride)
{
  constexpr int TPW = COUTB / 16;   // WMMA tiles per wave
  extern __shared__ unsigned short smem_us[];
  unsigned short* Xs  = smem_us;                                   // [Cin][128] bf16
  unsigned short* Bc0 = (TAPS == 9) ? (smem_us + Cin * 128) : smem_us; // [128][40]
  unsigned short* Bc1 = Bc0 + 128 * 40;                            // second buffer

  const int tid  = threadIdx.x;
  const int poly = blockIdx.x;
  const int mrowBase = blockIdx.y * COUTB;
  const unsigned short* Xg = Xbase + (size_t)poly * xPolyStride;

  if (TAPS == 9) {
    int total = Cin * 128;
    for (int idx = tid * 8; idx < total; idx += 256 * 8)
      *(int4*)(Xs + idx) = *(const int4*)(Xg + idx);
  }

  const int wave  = tid >> 5;
  const int lane  = tid & 31;
  const int mloc  = lane & 15;
  const int khalf = lane >> 4;
  const int K     = Cin * TAPS;
  const int KC    = Kpad >> 5;

  // mt is constant per wave; nt covers a contiguous run of TPW tiles.
  const int mtC    = (wave * TPW) >> 3;
  const int ntBase = (wave * TPW) & 7;
  const unsigned short* wrowBase =
      Wb + (size_t)(mrowBase + mtC * 16 + mloc) * Kpad;

  // Branchless, pair-packed panel build: [n][k] transposed panel, b32 stores.
  auto buildPanel = [&](int kc, unsigned short* Bp) {
    #pragma unroll
    for (int it = 0; it < 8; ++it) {
      int idx = tid + it * 256;            // 0..2047 (pairs)
      int n  = idx & 127;
      int kp = idx >> 7;                   // 0..15
      int k0 = (kc << 5) + kp * 2;
      unsigned int v0, v1;
      if (TAPS == 9) {
        int c0 = k0 / 9,      t0 = k0 - c0 * 9;
        int c1 = (k0 + 1) / 9, t1 = (k0 + 1) - c1 * 9;
        int p0 = (n + (t0 - 4) * dil + 128) & 127;   // circular pad
        int p1 = (n + (t1 - 4) * dil + 128) & 127;
        v0 = Xs[min(c0, Cin - 1) * 128 + p0];
        v1 = Xs[min(c1, Cin - 1) * 128 + p1];
      } else {
        v0 = Xg[(size_t)min(k0, K - 1) * 128 + n];
        v1 = Xg[(size_t)min(k0 + 1, K - 1) * 128 + n];
      }
      v0 = (k0     < K) ? v0 : 0u;
      v1 = (k0 + 1 < K) ? v1 : 0u;
      *(unsigned int*)(Bp + n * 40 + kp * 2) = v0 | (v1 << 16);
    }
  };
  auto loadA = [&](int kc) {
    FragU a;
    const unsigned short* wrow = wrowBase + (kc << 5);
    a.q[0] = *(const int4*)(wrow + khalf * 8);
    a.q[1] = *(const int4*)(wrow + 16 + khalf * 8);
    return a;
  };

  v8f acc[TPW];
  #pragma unroll
  for (int j = 0; j < TPW; ++j) {
    #pragma unroll
    for (int r = 0; r < 8; ++r) acc[j][r] = 0.0f;
  }

  buildPanel(0, Bc0);
  FragU aCur = loadA(0);
  __syncthreads();

  for (int kc = 0; kc < KC; ++kc) {
    unsigned short* Bp = (kc & 1) ? Bc1 : Bc0;
    unsigned short* Bn = (kc & 1) ? Bc0 : Bc1;

    FragU aNext = aCur;
    if (kc + 1 < KC) {
      buildPanel(kc + 1, Bn);          // overlaps with this chunk's WMMAs
      aNext = loadA(kc + 1);           // register-pipelined A fragment
    }
    // prefetch weight row two chunks ahead (global_prefetch_b8)
    __builtin_prefetch(wrowBase + (min(kc + 2, KC - 1) << 5), 0, 0);

    #pragma unroll
    for (int j = 0; j < TPW; ++j) {
      int nt = ntBase + j;
      FragU bF;
      const unsigned short* brow = Bp + (nt * 16 + mloc) * 40;
      bF.q[0] = *(const int4*)(brow + khalf * 8);
      bF.q[1] = *(const int4*)(brow + 16 + khalf * 8);
      acc[j] = __builtin_amdgcn_wmma_f32_16x16x32_bf16(
          false, aCur.v, false, bF.v, (short)0, acc[j], false, false);
    }
    aCur = aNext;
    __syncthreads();                   // single barrier per chunk
  }

  // Epilogue: bias / extra / relu / bn / residual, round to bf16.
  #pragma unroll
  for (int j = 0; j < TPW; ++j) {
    int nt = ntBase + j;
    int n = nt * 16 + mloc;
    #pragma unroll
    for (int r = 0; r < 8; ++r) {
      int mloc2 = mtC * 16 + r + 8 * khalf;
      int m = mrowBase + mloc2;
      float val = acc[j][r] + bias[m];
      if (flags & 8) val += extra[(size_t)poly * extraStride + m];
      if (flags & 1) val = fmaxf(val, 0.0f);
      if (flags & 2)
        val = (val - bnm[m]) * rsqrtf(bnv[m] + EPSN) * bng[m] + bnb[m];
      if (TAPS == 9) { if (flags & 4) val += bf2f(Xs[mloc2 * 128 + n]); }
      Out[(size_t)poly * outPolyStride + (size_t)m * 128 + n] = f2bf(val);
    }
  }
}

// ---------------------------------------------------------------------------
// Weight prep (fp32 -> bf16, with padding / slicing)
// ---------------------------------------------------------------------------
__global__ void cvt_bf16_kernel(const float* __restrict__ s,
                                unsigned short* __restrict__ d, int n) {
  int i = blockIdx.x * 256 + threadIdx.x;
  if (i < n) d[i] = f2bf(s[i]);
}
__global__ void headpad_kernel(const float* __restrict__ s,
                               unsigned short* __restrict__ d) {
  int i = blockIdx.x * 256 + threadIdx.x;
  if (i < 128 * 608) {
    int m = i / 608, k = i - m * 608;
    d[i] = (k < 594) ? f2bf(s[m * 594 + k]) : (unsigned short)0;
  }
}
__global__ void pr1b_kernel(const float* __restrict__ s,
                            unsigned short* __restrict__ d) {
  int i = blockIdx.x * 256 + threadIdx.x;
  if (i < 256 * 1024) {
    int m = i >> 10, k = i & 1023;
    d[i] = f2bf(s[m * 1280 + 256 + k]);
  }
}

// ---------------------------------------------------------------------------
// edge_predict: per-pixel 1x1 64->32 (BN,relu) -> 1; also s = 1 - sigmoid(e)
// ---------------------------------------------------------------------------
__global__ __launch_bounds__(256)
void edge_predict_kernel(const float* __restrict__ feat,
                         const float* __restrict__ w1,
                         const float* __restrict__ bng, const float* __restrict__ bnb,
                         const float* __restrict__ bnm, const float* __restrict__ bnv,
                         const float* __restrict__ w2,
                         float* __restrict__ edgeOut, float* __restrict__ sOut) {
  __shared__ float W1s[2048];
  __shared__ float W2s[32], Gs[32], Bs[32], Ms[32], Vs[32];
  for (int i = threadIdx.x; i < 2048; i += 256) W1s[i] = w1[i];
  if (threadIdx.x < 32) {
    int o = threadIdx.x;
    W2s[o] = w2[o]; Gs[o] = bng[o]; Bs[o] = bnb[o]; Ms[o] = bnm[o]; Vs[o] = bnv[o];
  }
  __syncthreads();
  int q = blockIdx.x * 256 + threadIdx.x;     // 4*256*256 pixels
  int b = q >> 16, hw = q & 65535;
  const float* fp = feat + (size_t)b * 64 * 65536 + hw;
  float h[32];
  #pragma unroll
  for (int o = 0; o < 32; ++o) h[o] = 0.f;
  for (int c = 0; c < 64; ++c) {
    float f = fp[(size_t)c * 65536];
    #pragma unroll
    for (int o = 0; o < 32; ++o) h[o] += f * W1s[o * 64 + c];
  }
  float e = 0.f;
  #pragma unroll
  for (int o = 0; o < 32; ++o) {
    float y = (h[o] - Ms[o]) * rsqrtf(Vs[o] + EPSN) * Gs[o] + Bs[o];
    y = fmaxf(y, 0.f);
    e += y * W2s[o];
  }
  edgeOut[q] = e;
  sOut[q] = 1.f - 1.f / (1.f + __expf(-e));
}

// ---------------------------------------------------------------------------
// edge_att 3x3 convs + GroupNorm(4)
// ---------------------------------------------------------------------------
__global__ __launch_bounds__(256)
void ea_conv1_kernel(const float* __restrict__ s, const float* __restrict__ w,
                     float* __restrict__ out) {
  __shared__ float Ws[288];
  for (int i = threadIdx.x; i < 288; i += 256) Ws[i] = w[i];
  __syncthreads();
  int idx = blockIdx.x * 256 + threadIdx.x;   // 4*32*65536
  int x = idx & 255, y = (idx >> 8) & 255, c = (idx >> 16) & 31, b = idx >> 21;
  const float* sp = s + (size_t)b * 65536;
  float a = 0.f;
  #pragma unroll
  for (int dy = -1; dy <= 1; ++dy) {
    int yy = y + dy; if ((unsigned)yy > 255u) continue;
    #pragma unroll
    for (int dx = -1; dx <= 1; ++dx) {
      int xx = x + dx; if ((unsigned)xx > 255u) continue;
      a += sp[yy * 256 + xx] * Ws[c * 9 + (dy + 1) * 3 + dx + 1];
    }
  }
  out[idx] = a;
}

__global__ __launch_bounds__(256)
void ea_conv2_kernel(const float* __restrict__ t, const float* __restrict__ w,
                     float* __restrict__ out) {
  __shared__ float Ws[9216];
  for (int i = threadIdx.x; i < 9216; i += 256) Ws[i] = w[i];
  __syncthreads();
  int idx = blockIdx.x * 256 + threadIdx.x;
  int x = idx & 255, y = (idx >> 8) & 255, o = (idx >> 16) & 31, b = idx >> 21;
  const float* tp = t + (size_t)b * 32 * 65536;
  float a = 0.f;
  for (int c = 0; c < 32; ++c) {
    const float* tc = tp + (size_t)c * 65536;
    const float* wc = Ws + (o * 32 + c) * 9;
    #pragma unroll
    for (int dy = -1; dy <= 1; ++dy) {
      int yy = y + dy; if ((unsigned)yy > 255u) continue;
      #pragma unroll
      for (int dx = -1; dx <= 1; ++dx) {
        int xx = x + dx; if ((unsigned)xx > 255u) continue;
        a += tc[yy * 256 + xx] * wc[(dy + 1) * 3 + dx + 1];
      }
    }
  }
  out[idx] = a;
}

__global__ __launch_bounds__(256)
void ea_conv3_kernel(const float* __restrict__ t, const float* __restrict__ w,
                     float* __restrict__ out) {
  __shared__ float Ws[288];
  for (int i = threadIdx.x; i < 288; i += 256) Ws[i] = w[i];
  __syncthreads();
  int idx = blockIdx.x * 256 + threadIdx.x;   // 4*65536
  int x = idx & 255, y = (idx >> 8) & 255, b = idx >> 16;
  const float* tp = t + (size_t)b * 32 * 65536;
  float a = 0.f;
  for (int c = 0; c < 32; ++c) {
    const float* tc = tp + (size_t)c * 65536;
    const float* wc = Ws + c * 9;
    #pragma unroll
    for (int dy = -1; dy <= 1; ++dy) {
      int yy = y + dy; if ((unsigned)yy > 255u) continue;
      #pragma unroll
      for (int dx = -1; dx <= 1; ++dx) {
        int xx = x + dx; if ((unsigned)xx > 255u) continue;
        a += tc[yy * 256 + xx] * wc[(dy + 1) * 3 + dx + 1];
      }
    }
  }
  out[idx] = 1.f / (1.f + __expf(-a));
}

__global__ void gn_zero_kernel(float* st) {
  if (threadIdx.x < 32) st[threadIdx.x] = 0.f;
}
__global__ __launch_bounds__(256)
void gn_stats_kernel(const float* __restrict__ x, float* __restrict__ st) {
  __shared__ float s1[256], s2[256];
  int bg = blockIdx.y;     // b*4 + group (channels contiguous)
  const float* xp = x + (size_t)bg * 8 * 65536 + (size_t)blockIdx.x * 8192;
  float a = 0.f, b = 0.f;
  for (int i = threadIdx.x; i < 8192; i += 256) { float v = xp[i]; a += v; b += v * v; }
  s1[threadIdx.x] = a; s2[threadIdx.x] = b;
  __syncthreads();
  for (int s = 128; s > 0; s >>= 1) {
    if (threadIdx.x < s) { s1[threadIdx.x] += s1[threadIdx.x + s]; s2[threadIdx.x] += s2[threadIdx.x + s]; }
    __syncthreads();
  }
  if (threadIdx.x == 0) { atomicAdd(&st[bg * 2], s1[0]); atomicAdd(&st[bg * 2 + 1], s2[0]); }
}
__global__ __launch_bounds__(256)
void gn_apply_kernel(float* __restrict__ x, const float* __restrict__ st,
                     const float* __restrict__ g, const float* __restrict__ b) {
  int idx = blockIdx.x * 256 + threadIdx.x;
  int c = (idx >> 16) & 31, bb = idx >> 21;
  int bg = bb * 4 + (c >> 3);
  float mean = st[bg * 2] * (1.f / 524288.f);
  float var  = st[bg * 2 + 1] * (1.f / 524288.f) - mean * mean;
  float v = (x[idx] - mean) * rsqrtf(var + EPSN) * g[c] + b[c];
  x[idx] = fmaxf(v, 0.f);
}

// ---------------------------------------------------------------------------
// Bilinear grid-sample gather -> snake input X0 [n][66][128] bf16 + edge_poly
// ---------------------------------------------------------------------------
__global__ __launch_bounds__(128)
void sample_kernel(const float* __restrict__ feat, const float* __restrict__ efeat,
                   const float* __restrict__ poly, const float* __restrict__ can,
                   const int* __restrict__ ind,
                   unsigned short* __restrict__ X0, float* __restrict__ edgePoly) {
  int n = blockIdx.x, p = threadIdx.x;
  float px = poly[((size_t)n * 128 + p) * 2 + 0];
  float py = poly[((size_t)n * 128 + p) * 2 + 1];
  int b = ind[n];
  float gx = px / 128.f - 1.f, gy = py / 128.f - 1.f;
  float xs = ((gx + 1.f) * 256.f - 1.f) * 0.5f;
  float ys = ((gy + 1.f) * 256.f - 1.f) * 0.5f;
  float x0f = floorf(xs), y0f = floorf(ys);
  float wx = xs - x0f, wy = ys - y0f;
  int x0 = (int)x0f, y0 = (int)y0f, x1 = x0 + 1, y1 = y0 + 1;
  float v00 = (x0 >= 0 && x0 < 256 && y0 >= 0 && y0 < 256) ? 1.f : 0.f;
  float v01 = (x1 >= 0 && x1 < 256 && y0 >= 0 && y0 < 256) ? 1.f : 0.f;
  float v10 = (x0 >= 0 && x0 < 256 && y1 >= 0 && y1 < 256) ? 1.f : 0.f;
  float v11 = (x1 >= 0 && x1 < 256 && y1 >= 0 && y1 < 256) ? 1.f : 0.f;
  float w00 = (1.f - wx) * (1.f - wy) * v00;
  float w01 = wx * (1.f - wy) * v01;
  float w10 = (1.f - wx) * wy * v10;
  float w11 = wx * wy * v11;
  int cx0 = min(max(x0, 0), 255), cx1 = min(max(x1, 0), 255);
  int cy0 = min(max(y0, 0), 255), cy1 = min(max(y1, 0), 255);
  const float* fb = feat + (size_t)b * 64 * 65536;
  unsigned short* xo = X0 + (size_t)n * 66 * 128;
  for (int c = 0; c < 64; ++c) {
    const float* fc = fb + (size_t)c * 65536;
    float v = fc[cy0 * 256 + cx0] * w00 + fc[cy0 * 256 + cx1] * w01 +
              fc[cy1 * 256 + cx0] * w10 + fc[cy1 * 256 + cx1] * w11;
    xo[c * 128 + p] = f2bf(v);
  }
  xo[64 * 128 + p] = f2bf(can[((size_t)n * 128 + p) * 2 + 0] * 4.0f);
  xo[65 * 128 + p] = f2bf(can[((size_t)n * 128 + p) * 2 + 1] * 4.0f);
  const float* eb = efeat + (size_t)b * 65536;
  float ev = eb[cy0 * 256 + cx0] * w00 + eb[cy0 * 256 + cx1] * w01 +
             eb[cy1 * 256 + cx0] * w10 + eb[cy1 * 256 + cx1] * w11;
  edgePoly[(size_t)n * 128 + p] = ev;
}

// ---------------------------------------------------------------------------
// global max over points -> gs ; v = pr_w1[:, :256] @ gs
// ---------------------------------------------------------------------------
__global__ __launch_bounds__(256)
void gs_kernel(const unsigned short* __restrict__ fused, float* __restrict__ gs) {
  int id = blockIdx.x * 256 + threadIdx.x;        // 512*256
  int n = id >> 8, m = id & 255;
  const unsigned short* fp = fused + ((size_t)n * 256 + m) * 128;
  float mx = -3.4e38f;
  for (int p = 0; p < 128; ++p) mx = fmaxf(mx, bf2f(fp[p]));
  gs[id] = mx;
}
__global__ __launch_bounds__(256)
void gemv_kernel(const float* __restrict__ w1, const float* __restrict__ gs,
                 float* __restrict__ vout) {
  int id = blockIdx.x * 256 + threadIdx.x;        // 512*256
  int n = id >> 8, m = id & 255;
  const float* wr = w1 + (size_t)m * 1280;        // cols 0..255 multiply gs
  const float* g = gs + (size_t)n * 256;
  float a = 0.f;
  for (int c = 0; c < 256; ++c) a += wr[c] * g[c];
  vout[id] = a;
}

// ---------------------------------------------------------------------------
// Final 2x64 head + poly update + per-polygon min reduction (can poly)
// ---------------------------------------------------------------------------
__global__ __launch_bounds__(128)
void final_kernel(const unsigned short* __restrict__ y2,
                  const float* __restrict__ w3, const float* __restrict__ b3,
                  const float* __restrict__ edgePoly, const float* __restrict__ polyIn,
                  float* __restrict__ pysOut, float* __restrict__ curPoly,
                  float* __restrict__ canPoly) {
  __shared__ float mxs[128], mys[128];
  int n = blockIdx.x, p = threadIdx.x;
  const unsigned short* yp = y2 + (size_t)n * 64 * 128;
  float o0 = b3[0], o1 = b3[1];
  for (int c = 0; c < 64; ++c) {
    float v = bf2f(yp[c * 128 + p]);
    o0 += w3[c] * v;
    o1 += w3[64 + c] * v;
  }
  float e = edgePoly[(size_t)n * 128 + p];
  float px = polyIn[((size_t)n * 128 + p) * 2 + 0] * 4.0f + o0 * e;
  float py = polyIn[((size_t)n * 128 + p) * 2 + 1] * 4.0f + o1 * e;
  pysOut[((size_t)n * 128 + p) * 2 + 0] = px;
  pysOut[((size_t)n * 128 + p) * 2 + 1] = py;
  float cx = px * 0.25f, cy = py * 0.25f;
  curPoly[((size_t)n * 128 + p) * 2 + 0] = cx;
  curPoly[((size_t)n * 128 + p) * 2 + 1] = cy;
  mxs[p] = cx; mys[p] = cy;
  __syncthreads();
  for (int s = 64; s > 0; s >>= 1) {
    if (p < s) { mxs[p] = fminf(mxs[p], mxs[p + s]); mys[p] = fminf(mys[p], mys[p + s]); }
    __syncthreads();
  }
  canPoly[((size_t)n * 128 + p) * 2 + 0] = cx - mxs[0];
  canPoly[((size_t)n * 128 + p) * 2 + 1] = cy - mys[0];
}

// ---------------------------------------------------------------------------
// Host driver
// ---------------------------------------------------------------------------
extern "C" void kernel_launch(void* const* d_in, const int* in_sizes, int n_in,
                              void* d_out, int out_size, void* d_ws, size_t ws_size,
                              hipStream_t stream) {
  (void)in_sizes; (void)n_in; (void)out_size; (void)ws_size;
  const float* cnn  = (const float*)d_in[0];
  const int*   ind  = (const int*)d_in[3];

  // params[i] pytree leaves (dict keys sorted, tuples/lists in order): 69 per iter
  auto P = [&](int it, int k) -> const float* {
    return (const float*)d_in[4 + it * 69 + k];
  };
  // 0 ea_gn1_b 1 ea_gn1_g 2 ea_gn2_b 3 ea_gn2_g 4 ea_w1 5 ea_w2 6 ea_w3
  // 7 ep_bn_b 8 ep_bn_g 9 ep_bn_m 10 ep_bn_v 11 ep_w1 12 ep_w2
  // 13 fus_b 14 fus_w 15 head_b 16-19 head_bn(g,b,m,v) 20 head_w
  // 21 pr_b1 22 pr_b2 23 pr_b3 24 pr_w1 25 pr_w2 26 pr_w3
  // 27+6r res_b  28..31+6r res_bn(g,b,m,v)  32+6r res_w

  char* base = (char*)d_ws;
  size_t off = 0;
  auto carve = [&](size_t bytes) -> void* {
    void* r = base + off;
    off += (bytes + 255) & ~(size_t)255;
    return r;
  };

  unsigned short *wHead[3], *wRes[3][7], *wFus[3], *wPr1b[3], *wPr2[3];
  for (int i = 0; i < 3; ++i) {
    wHead[i] = (unsigned short*)carve((size_t)128 * 608 * 2);
    for (int r = 0; r < 7; ++r) wRes[i][r] = (unsigned short*)carve((size_t)128 * 1152 * 2);
    wFus[i]  = (unsigned short*)carve((size_t)256 * 1024 * 2);
    wPr1b[i] = (unsigned short*)carve((size_t)256 * 1024 * 2);
    wPr2[i]  = (unsigned short*)carve((size_t)64 * 256 * 2);
  }
  float* sbuf   = (float*)carve((size_t)4 * 65536 * 4);
  float* t1     = (float*)carve((size_t)4 * 32 * 65536 * 4);
  float* t2     = (float*)carve((size_t)4 * 32 * 65536 * 4);
  float* efeat  = (float*)carve((size_t)4 * 65536 * 4);
  float* gnst   = (float*)carve(32 * 4);
  unsigned short* X0    = (unsigned short*)carve((size_t)512 * 66 * 128 * 2);
  unsigned short* state = (unsigned short*)carve((size_t)512 * 1024 * 128 * 2);
  unsigned short* fused = (unsigned short*)carve((size_t)512 * 256 * 128 * 2);
  unsigned short* y1b   = (unsigned short*)carve((size_t)512 * 256 * 128 * 2);
  unsigned short* y2b   = (unsigned short*)carve((size_t)512 * 64 * 128 * 2);
  float* gsb   = (float*)carve((size_t)512 * 256 * 4);
  float* vb    = (float*)carve((size_t)512 * 256 * 4);
  float* ePoly = (float*)carve((size_t)512 * 128 * 4);
  float* curP[2] = { (float*)carve((size_t)512 * 128 * 2 * 4),
                     (float*)carve((size_t)512 * 128 * 2 * 4) };
  float* canP[2] = { (float*)carve((size_t)512 * 128 * 2 * 4),
                     (float*)carve((size_t)512 * 128 * 2 * 4) };

  // --- weight prep (fp32 -> bf16, every call; deterministic) ---
  for (int i = 0; i < 3; ++i) {
    headpad_kernel<<<(128 * 608 + 255) / 256, 256, 0, stream>>>(P(i, 20), wHead[i]);
    for (int r = 0; r < 7; ++r)
      cvt_bf16_kernel<<<(147456 + 255) / 256, 256, 0, stream>>>(P(i, 32 + 6 * r), wRes[i][r], 147456);
    cvt_bf16_kernel<<<(262144 + 255) / 256, 256, 0, stream>>>(P(i, 14), wFus[i], 262144);
    pr1b_kernel<<<(262144 + 255) / 256, 256, 0, stream>>>(P(i, 24), wPr1b[i]);
    cvt_bf16_kernel<<<(16384 + 255) / 256, 256, 0, stream>>>(P(i, 25), wPr2[i], 16384);
  }

  float* outF = (float*)d_out;
  const int dils[7] = {1, 1, 1, 2, 2, 4, 4};
  const int PANELS = 2 * 128 * 40 * 2;   // double-buffered im2col panel bytes

  for (int i = 0; i < 3; ++i) {
    float* pysOut  = outF + (size_t)i * 131072;
    float* edgeOut = outF + 393216 + (size_t)i * 262144;
    const float* polyIn = (i == 0) ? (const float*)d_in[1] : curP[(i - 1) & 1];
    const float* canIn  = (i == 0) ? (const float*)d_in[2] : canP[(i - 1) & 1];

    // edge_predict + s = 1 - sigmoid(edge)
    edge_predict_kernel<<<1024, 256, 0, stream>>>(
        cnn, P(i, 11), P(i, 8), P(i, 7), P(i, 9), P(i, 10), P(i, 12), edgeOut, sbuf);

    // edge_att
    ea_conv1_kernel<<<32768, 256, 0, stream>>>(sbuf, P(i, 4), t1);
    gn_zero_kernel<<<1, 32, 0, stream>>>(gnst);
    gn_stats_kernel<<<dim3(64, 16), 256, 0, stream>>>(t1, gnst);
    gn_apply_kernel<<<32768, 256, 0, stream>>>(t1, gnst, P(i, 1), P(i, 0));
    ea_conv2_kernel<<<32768, 256, 0, stream>>>(t1, P(i, 5), t2);
    gn_zero_kernel<<<1, 32, 0, stream>>>(gnst);
    gn_stats_kernel<<<dim3(64, 16), 256, 0, stream>>>(t2, gnst);
    gn_apply_kernel<<<32768, 256, 0, stream>>>(t2, gnst, P(i, 3), P(i, 2));
    ea_conv3_kernel<<<1024, 256, 0, stream>>>(t2, P(i, 6), efeat);

    // gather features at polygon points
    sample_kernel<<<512, 128, 0, stream>>>(cnn, efeat, polyIn, canIn, ind, X0, ePoly);

    // snake head: 66 -> 128, k=9, d=1   (WMMA bf16)
    snake_gemm<128, 9><<<dim3(512, 1), 256, 66 * 128 * 2 + PANELS, stream>>>(
        X0, 66 * 128, 66, 1, 608, wHead[i], P(i, 15),
        P(i, 16), P(i, 17), P(i, 18), P(i, 19),
        nullptr, 0, /*relu|bn*/ 3, state, 131072);

    // 7 residual blocks
    for (int r = 0; r < 7; ++r) {
      snake_gemm<128, 9><<<dim3(512, 1), 256, 128 * 128 * 2 + PANELS, stream>>>(
          state + r * 16384, 131072, 128, dils[r], 1152, wRes[i][r], P(i, 27 + 6 * r),
          P(i, 28 + 6 * r), P(i, 29 + 6 * r), P(i, 30 + 6 * r), P(i, 31 + 6 * r),
          nullptr, 0, /*relu|bn|res*/ 7, state + (r + 1) * 16384, 131072);
    }

    // fusion 1024 -> 256 (bias only)
    snake_gemm<128, 1><<<dim3(512, 2), 256, PANELS, stream>>>(
        state, 131072, 1024, 1, 1024, wFus[i], P(i, 13),
        nullptr, nullptr, nullptr, nullptr, nullptr, 0, 0, fused, 32768);

    gs_kernel<<<512, 256, 0, stream>>>(fused, gsb);
    gemv_kernel<<<512, 256, 0, stream>>>(P(i, 24), gsb, vb);

    // prediction layer 1: state part + per-poly gs vector, relu
    snake_gemm<128, 1><<<dim3(512, 2), 256, PANELS, stream>>>(
        state, 131072, 1024, 1, 1024, wPr1b[i], P(i, 21),
        nullptr, nullptr, nullptr, nullptr, vb, 256, /*relu|extra*/ 9, y1b, 32768);

    // prediction layer 2: 256 -> 64, relu
    snake_gemm<64, 1><<<dim3(512, 1), 256, PANELS, stream>>>(
        y1b, 32768, 256, 1, 256, wPr2[i], P(i, 22),
        nullptr, nullptr, nullptr, nullptr, nullptr, 0, 1, y2b, 8192);

    // final 2x64 head, poly update, min reduction for can-poly
    final_kernel<<<512, 128, 0, stream>>>(
        y2b, P(i, 26), P(i, 23), ePoly, polyIn, pysOut, curP[i & 1], canP[i & 1]);
  }
}
